// NHeadAttention_57423712748006
// MI455X (gfx1250) — matrix operand order repaired and checked by reading, not compiled, not run on hardware
//
#include <hip/hip_runtime.h>
#include <hip/hip_bf16.h>

// ---------------------------------------------------------------------------
// NHeadAttention for MI455X (gfx1250, wave32, WMMA).
//   B=2, S=4096, E=512, H=8, D=64.
//   k0: f32 -> f16 conversion of x and Wq/Wk/Wv (1/sqrt(D) folded into Wq).
//       Memory-bound pre-pass (~30MB total) so the GEMM loops carry zero
//       conversion VALU work.
//   k1: QKV projection: 32 tokens x 64 features per wave, pure b128 loads +
//       v_wmma_f32_16x16x32_f16. Q/K stored [B,H,S,D] f16; V stored [B,H,D,S].
//   k2: flash attention, 32 query rows per wave (two 16-row tiles sharing the
//       same K/V fragments -> 16 WMMAs per 8KB of operand traffic).
//       Score tiles computed transposed (S^T = K * Q^T) so softmax stats are
//       per-lane and P remaps lane-for-lane into the next GEMM's A fragment.
//   k3: LayerNorm over E=512.
// Workspace: Qh 8MB | Kh 8MB | V^T 8MB | ctx 16MB | xh 8MB | Wh 3x0.5MB
// ---------------------------------------------------------------------------

typedef _Float16 h16;
typedef h16   v16h __attribute__((ext_vector_type(16)));
typedef float v8f  __attribute__((ext_vector_type(8)));

union Frag {
    v16h  v;
    h16   e[16];
    uint4 q[2];
};

#define WMMA(a, b, c) \
    __builtin_amdgcn_wmma_f32_16x16x32_f16(false, (a), false, (b), (short)0, (c), false, false)

#define NB   2
#define NS   4096
#define NE   512
#define NH   8
#define ND   64

// ---------------------------------------------------------------------------
// Kernel 0: elementwise f32 -> f16 with scale (vectorized x4).
// ---------------------------------------------------------------------------
__global__ __launch_bounds__(256) void cvt_kernel(const float* __restrict__ src,
                                                  h16* __restrict__ dst,
                                                  int n, float scale)
{
    const int i = (blockIdx.x * blockDim.x + threadIdx.x) * 4;
    if (i < n) {
        const float4 v = *(const float4*)(src + i);
        h16 o[4] = { (h16)(v.x * scale), (h16)(v.y * scale),
                     (h16)(v.z * scale), (h16)(v.w * scale) };
        *(uint2*)(dst + i) = *(const uint2*)o;
    }
}

// ---------------------------------------------------------------------------
// Kernel 1: y = x @ W^T + b for W in {Wq,Wk,Wv} (all pre-converted f16).
// Each wave computes a 32(token) x 64(out-feature) strip: two A fragments
// share every B fragment -> 8 WMMAs per K-step.
// ---------------------------------------------------------------------------
__global__ __launch_bounds__(256) void qkv_kernel(
    const h16* __restrict__ xh,
    const h16* __restrict__ Wqh, const float* __restrict__ bq,
    const h16* __restrict__ Wkh, const float* __restrict__ bk,
    const h16* __restrict__ Wvh, const float* __restrict__ bv,
    h16* __restrict__ Qh, h16* __restrict__ Kh, h16* __restrict__ Vt)
{
    const int lane = threadIdx.x & 31;
    const int wave = (blockIdx.x * blockDim.x + threadIdx.x) >> 5;
    const int which = wave / ((NB * NS / 32) * (NE / 64));   // 0=Q 1=K 2=V
    const int rem   = wave % ((NB * NS / 32) * (NE / 64));
    const int M0    = (rem >> 3) << 5;     // token block base (32 rows)
    const int N0    = (rem & 7)  << 6;     // out-feature base (64 cols)
    const int lo = lane & 15, hi = lane >> 4;

    const h16*   W    = (which == 0) ? Wqh : (which == 1) ? Wkh : Wvh;
    const float* bias = (which == 0) ? bq  : (which == 1) ? bk  : bv;

    v8f acc0[4] = {}, acc1[4] = {};

    const h16* xr0 = xh + (size_t)(M0 + lo) * NE;
    const h16* xr1 = xr0 + 16 * NE;
    for (int kk = 0; kk < NE; kk += 32) {
        // A fragments: lane = row m (=lo), halfs j<8 -> k=8hi+j, j>=8 -> k=8hi+16+(j-8)
        Frag a0, a1;
        a0.q[0] = *(const uint4*)(xr0 + kk + 8 * hi);
        a0.q[1] = *(const uint4*)(xr0 + kk + 8 * hi + 16);
        a1.q[0] = *(const uint4*)(xr1 + kk + 8 * hi);
        a1.q[1] = *(const uint4*)(xr1 + kk + 8 * hi + 16);
        #pragma unroll
        for (int f = 0; f < 4; ++f) {
            // B fragment: lane = col n = N0+16f+lo, j<8 -> k=8hi+j, j>=8 -> k=8hi+16+(j-8)
            Frag bw;
            const h16* wb = W + (size_t)(N0 + 16 * f + lo) * NE + kk + 8 * hi;
            bw.q[0] = *(const uint4*)(wb);
            bw.q[1] = *(const uint4*)(wb + 16);
            acc0[f] = WMMA(a0.v, bw.v, acc0[f]);
            acc1[f] = WMMA(a1.v, bw.v, acc1[f]);
        }
    }

    const int h = N0 >> 6;   // head (N0 is a multiple of 64, strip width 64)
    #pragma unroll
    for (int f = 0; f < 4; ++f) {
        const int o = N0 + 16 * f + lo;
        const int d = 16 * f + lo;
        // Wq carries the 1/sqrt(D) scale already; apply it to bq here too.
        const float bb = bias[o] * ((which == 0) ? 0.125f : 1.0f);
        #pragma unroll
        for (int r = 0; r < 8; ++r) {
            #pragma unroll
            for (int half = 0; half < 2; ++half) {
                const int tok = M0 + 16 * half + r + 8 * hi;
                const int b = tok >> 12, s = tok & (NS - 1);
                const float val = (half ? acc1[f][r] : acc0[f][r]) + bb;
                const h16 hv = (h16)val;
                const size_t bhp = (size_t)(b * NH + h);
                if (which == 2)      Vt[(bhp * ND + d) * NS + s] = hv;   // transposed
                else if (which == 0) Qh[(bhp * NS + s) * ND + d] = hv;
                else                 Kh[(bhp * NS + s) * ND + d] = hv;
            }
        }
    }
}

// ---------------------------------------------------------------------------
// Online-softmax update for one 16-row query tile held transposed in two
// score C-tiles (st0: t in [T,T+16), st1: t in [T+16,T+32)). Per lane the
// query row is s = lo; lane^16 holds the other 16 key positions.
// Produces the P A-fragment for the P*V GEMM (in-lane remap) and the rescale
// factor alpha for the running context accumulators.
// ---------------------------------------------------------------------------
__device__ __forceinline__ void softmax_update(const v8f& st0, const v8f& st1,
                                               float& mrow, float& lrow,
                                               Frag& pa, float& alpha)
{
    float tmax = st0[0];
    #pragma unroll
    for (int r = 1; r < 8; ++r) tmax = fmaxf(tmax, st0[r]);
    #pragma unroll
    for (int r = 0; r < 8; ++r) tmax = fmaxf(tmax, st1[r]);
    tmax = fmaxf(tmax, __shfl_xor(tmax, 16));
    const float mnew = fmaxf(mrow, tmax);
    alpha = __expf(mrow - mnew);
    float psum = 0.0f;
    #pragma unroll
    for (int r = 0; r < 8; ++r) { float p = __expf(st0[r] - mnew); psum += p; pa.e[r]     = (h16)p; }
    #pragma unroll
    for (int r = 0; r < 8; ++r) { float p = __expf(st1[r] - mnew); psum += p; pa.e[8 + r] = (h16)p; }
    psum += __shfl_xor(psum, 16);
    lrow = lrow * alpha + psum;
    mrow = mnew;
}

// ---------------------------------------------------------------------------
// Kernel 2: flash attention. One wave owns 32 query rows (two 16-row tiles)
// of one (b,h) and streams all 4096 keys in blocks of 32. K/V fragments are
// shared by both query tiles: 16 WMMAs per iteration against 16 b128 loads.
// ---------------------------------------------------------------------------
__global__ __launch_bounds__(256) void attn_kernel(
    const h16* __restrict__ Qh, const h16* __restrict__ Kh,
    const h16* __restrict__ Vt, float* __restrict__ ctx)
{
    const int lane = threadIdx.x & 31;
    const int wave = (blockIdx.x * blockDim.x + threadIdx.x) >> 5;
    const int bh = wave >> 7;               // 16 (b,h) pairs, 128 waves each
    const int S0 = (wave & 127) << 5;       // 32-row query block
    const int lo = lane & 15, hi = lane >> 4;

    const h16* Qp = Qh + (size_t)bh * NS * ND;
    const h16* Kp = Kh + (size_t)bh * NS * ND;
    const h16* Vp = Vt + (size_t)bh * ND * NS;

    // Persistent Q^T B-fragments, two 16-row tiles x two d-halves.
    // Lane = col s (= lo within tile), j<8 -> k=8hi+j, j>=8 -> k=8hi+16+(j-8).
    Frag q0a, q0b, q1a, q1b;
    {
        const h16* qr = Qp + (size_t)(S0 + lo) * ND + 8 * hi;
        q0a.q[0] = *(const uint4*)(qr);      q0a.q[1] = *(const uint4*)(qr + 16);
        q0b.q[0] = *(const uint4*)(qr + 32); q0b.q[1] = *(const uint4*)(qr + 48);
        qr += 16 * ND;
        q1a.q[0] = *(const uint4*)(qr);      q1a.q[1] = *(const uint4*)(qr + 16);
        q1b.q[0] = *(const uint4*)(qr + 32); q1b.q[1] = *(const uint4*)(qr + 48);
    }

    v8f   cacc0[4] = {}, cacc1[4] = {};
    float m0 = -3.0e38f, l0 = 0.0f;
    float m1 = -3.0e38f, l1 = 0.0f;

    for (int T = 0; T < NS; T += 32) {
        // --- K A-fragments: two t-tiles x two d-halves (distinct regs for ILP)
        Frag ka0, ka1, kb0, kb1;
        {
            const h16* kr = Kp + (size_t)(T + lo) * ND + 8 * hi;
            ka0.q[0] = *(const uint4*)(kr);      ka0.q[1] = *(const uint4*)(kr + 16);
            ka1.q[0] = *(const uint4*)(kr + 32); ka1.q[1] = *(const uint4*)(kr + 48);
            kr += 16 * ND;
            kb0.q[0] = *(const uint4*)(kr);      kb0.q[1] = *(const uint4*)(kr + 16);
            kb1.q[0] = *(const uint4*)(kr + 32); kb1.q[1] = *(const uint4*)(kr + 48);
        }

        // --- V^T B-fragments (shared by both query tiles); issue loads early
        Frag vb0, vb1, vb2, vb3;
        {
            const h16* vr = Vp + (size_t)lo * NS + T + 8 * hi;
            vb0.q[0] = *(const uint4*)(vr);                 vb0.q[1] = *(const uint4*)(vr + 16);
            vb1.q[0] = *(const uint4*)(vr + 16 * NS);       vb1.q[1] = *(const uint4*)(vr + 16 * NS + 16);
            vb2.q[0] = *(const uint4*)(vr + 32 * NS);       vb2.q[1] = *(const uint4*)(vr + 32 * NS + 16);
            vb3.q[0] = *(const uint4*)(vr + 48 * NS);       vb3.q[1] = *(const uint4*)(vr + 48 * NS + 16);
        }

        // --- 8 score WMMAs, interleaved across 4 independent accumulators
        v8f st00 = {}, st01 = {}, st10 = {}, st11 = {};
        st00 = WMMA(ka0.v, q0a.v, st00);
        st01 = WMMA(kb0.v, q0a.v, st01);
        st10 = WMMA(ka0.v, q1a.v, st10);
        st11 = WMMA(kb0.v, q1a.v, st11);
        st00 = WMMA(ka1.v, q0b.v, st00);
        st01 = WMMA(kb1.v, q0b.v, st01);
        st10 = WMMA(ka1.v, q1b.v, st10);
        st11 = WMMA(kb1.v, q1b.v, st11);

        // --- two independent online-softmax chains
        Frag pa0, pa1;
        float a0, a1;
        softmax_update(st00, st01, m0, l0, pa0, a0);
        softmax_update(st10, st11, m1, l1, pa1, a1);

        // --- rescale accumulators (ctx lane: dv=lo, row s = r+8hi)
        #pragma unroll
        for (int r = 0; r < 8; ++r) {
            const float ar0 = __shfl(a0, r + 8 * hi);
            const float ar1 = __shfl(a1, r + 8 * hi);
            cacc0[0][r] *= ar0; cacc0[1][r] *= ar0; cacc0[2][r] *= ar0; cacc0[3][r] *= ar0;
            cacc1[0][r] *= ar1; cacc1[1][r] *= ar1; cacc1[2][r] *= ar1; cacc1[3][r] *= ar1;
        }

        // --- 8 PV WMMAs (independent accumulators)
        cacc0[0] = WMMA(pa0.v, vb0.v, cacc0[0]);
        cacc1[0] = WMMA(pa1.v, vb0.v, cacc1[0]);
        cacc0[1] = WMMA(pa0.v, vb1.v, cacc0[1]);
        cacc1[1] = WMMA(pa1.v, vb1.v, cacc1[1]);
        cacc0[2] = WMMA(pa0.v, vb2.v, cacc0[2]);
        cacc1[2] = WMMA(pa1.v, vb2.v, cacc1[2]);
        cacc0[3] = WMMA(pa0.v, vb3.v, cacc0[3]);
        cacc1[3] = WMMA(pa1.v, vb3.v, cacc1[3]);
    }

    // --- finalize: divide by row sums, write ctx[b][s][h*64+d] (f32)
    const int b = bh >> 3, h = bh & 7;
    float* cbase = ctx + ((size_t)b * NS) * NE + (size_t)h * ND + lo;
    #pragma unroll
    for (int r = 0; r < 8; ++r) {
        const float il0 = 1.0f / __shfl(l0, r + 8 * hi);
        const float il1 = 1.0f / __shfl(l1, r + 8 * hi);
        float* row0 = cbase + (size_t)(S0 + r + 8 * hi) * NE;
        float* row1 = cbase + (size_t)(S0 + 16 + r + 8 * hi) * NE;
        row0[0]  = cacc0[0][r] * il0;
        row0[16] = cacc0[1][r] * il0;
        row0[32] = cacc0[2][r] * il0;
        row0[48] = cacc0[3][r] * il0;
        row1[0]  = cacc1[0][r] * il1;
        row1[16] = cacc1[1][r] * il1;
        row1[32] = cacc1[2][r] * il1;
        row1[48] = cacc1[3][r] * il1;
    }
}

// ---------------------------------------------------------------------------
// Kernel 3: LayerNorm over E=512, one wave per token row.
// ---------------------------------------------------------------------------
__global__ __launch_bounds__(256) void ln_kernel(
    const float* __restrict__ ctx, const float* __restrict__ gamma,
    const float* __restrict__ beta, float* __restrict__ out)
{
    const int lane = threadIdx.x & 31;
    const int row  = (blockIdx.x * blockDim.x + threadIdx.x) >> 5;
    const float* p = ctx + (size_t)row * NE;
    float vals[16], s = 0.0f, ss = 0.0f;
    #pragma unroll
    for (int i = 0; i < 16; ++i) {
        const float v = p[lane + 32 * i];
        vals[i] = v; s += v; ss += v * v;
    }
    #pragma unroll
    for (int m = 16; m >= 1; m >>= 1) { s += __shfl_xor(s, m); ss += __shfl_xor(ss, m); }
    const float mu = s * (1.0f / NE);
    const float rs = rsqrtf(ss * (1.0f / NE) - mu * mu + 1e-5f);
    float* o = out + (size_t)row * NE;
    #pragma unroll
    for (int i = 0; i < 16; ++i) {
        const int e = lane + 32 * i;
        o[e] = (vals[i] - mu) * rs * gamma[e] + beta[e];
    }
}

// ---------------------------------------------------------------------------
extern "C" void kernel_launch(void* const* d_in, const int* in_sizes, int n_in,
                              void* d_out, int out_size, void* d_ws, size_t ws_size,
                              hipStream_t stream) {
    const float* x  = (const float*)d_in[0];
    const float* Wq = (const float*)d_in[1];
    const float* bq = (const float*)d_in[2];
    const float* Wk = (const float*)d_in[3];
    const float* bk = (const float*)d_in[4];
    const float* Wv = (const float*)d_in[5];
    const float* bv = (const float*)d_in[6];
    const float* gamma = (const float*)d_in[7];
    const float* beta  = (const float*)d_in[8];

    char* ws = (char*)d_ws;
    const size_t MB = 1u << 20;
    h16*   Qh  = (h16*)(ws);
    h16*   Kh  = (h16*)(ws + 8 * MB);
    h16*   Vt  = (h16*)(ws + 16 * MB);
    float* ctx = (float*)(ws + 24 * MB);
    h16*   xh  = (h16*)(ws + 40 * MB);
    h16*   Wqh = (h16*)(ws + 48 * MB);
    h16*   Wkh = (h16*)(ws + 48 * MB + 512 * 1024);
    h16*   Wvh = (h16*)(ws + 49 * MB);

    const int nx = NB * NS * NE;   // 4,194,304
    const int nw = NE * NE;        // 262,144
    cvt_kernel<<<nx / 4 / 256, 256, 0, stream>>>(x,  xh,  nx, 1.0f);
    cvt_kernel<<<nw / 4 / 256, 256, 0, stream>>>(Wq, Wqh, nw, 0.125f);  // fold 1/sqrt(D)
    cvt_kernel<<<nw / 4 / 256, 256, 0, stream>>>(Wk, Wkh, nw, 1.0f);
    cvt_kernel<<<nw / 4 / 256, 256, 0, stream>>>(Wv, Wvh, nw, 1.0f);

    // 3 * (8192/32) * (512/64) = 6144 waves -> 768 blocks of 8 waves
    qkv_kernel<<<768, 256, 0, stream>>>(xh, Wqh, bq, Wkh, bk, Wvh, bv, Qh, Kh, Vt);
    // 16 bh * 128 row-blocks (32 rows each) = 2048 waves -> 256 blocks
    attn_kernel<<<256, 256, 0, stream>>>(Qh, Kh, Vt, ctx);
    // 8192 rows -> 1024 blocks of 8 waves
    ln_kernel<<<1024, 256, 0, stream>>>(ctx, gamma, beta, (float*)d_out);
}
